// EncoderLayer_55241869361586
// MI455X (gfx1250) — matrix operand order repaired and checked by reading, not compiled, not run on hardware
//
#include <hip/hip_runtime.h>
#include <hip/hip_bf16.h>
#include <math.h>

// ---------------------------------------------------------------------------
// MI455X / gfx1250 implementation of the Informer-style encoder layer.
// Heavy GEMMs: v_wmma_f32_16x16x32_f16 (f16 in, f32 accumulate), 128x128 block
// tiles. Attention staging uses the Tensor Data Mover (tensor_load_to_lds +
// s_wait_tensorcnt). wave32, 256-thread blocks = 8 waves.
// ---------------------------------------------------------------------------

#define USE_TDM 1

typedef _Float16 half_t;
typedef __attribute__((ext_vector_type(16))) _Float16 v16h;
typedef __attribute__((ext_vector_type(8)))  _Float16 v8h;
typedef __attribute__((ext_vector_type(4)))  _Float16 v4h;
typedef __attribute__((ext_vector_type(8)))  float    v8f;
typedef __attribute__((ext_vector_type(4)))  float    v4f;
typedef __attribute__((ext_vector_type(4)))  unsigned int u32x4;
typedef __attribute__((ext_vector_type(8)))  int      i32x8;
typedef __attribute__((ext_vector_type(4)))  int      i32x4;

#define WMMA_F16(a, b, c) \
  __builtin_amdgcn_wmma_f32_16x16x32_f16(false, (a), false, (b), (short)0, (c), false, false)

static constexpr int Bc  = 4;
static constexpr int Lc  = 2048;
static constexpr int Dc  = 512;   // D_MODEL == H*D_K == H*D_V
static constexpr int Hc  = 8;
static constexpr int DKc = 64;
static constexpr int DFF = 2048;
static constexpr int SK  = 40;    // sample_k == u (top-k count)
static constexpr int UP  = 48;    // u padded to a multiple of 16 for WMMA
static constexpr int Mrows = Bc * Lc; // 8192

// ---------------------------------------------------------------------------
// Tensor Data Mover: async 2D f16 tile load Global -> LDS (CDNA5 ISA ch.8).
// Descriptor group0: count=1 | lds_addr | global_addr(57b) | type=2.
// Descriptor group1: data_size=2B, pad fields, tensor/tile dims, dim0 stride.
// pad_interval code: 4 -> 32 DWORDs, 6 -> 128 DWORDs; pad_amount code 3 -> 4 DW.
// ---------------------------------------------------------------------------
#if USE_TDM
__device__ __forceinline__ void tdm_load_2d_f16(
    unsigned lds_off_bytes, const void* gsrc,
    unsigned tile_w_elems, unsigned tile_rows, unsigned row_stride_elems,
    unsigned pad_interval_code, unsigned pad_amount_code) {
  unsigned long long ga = (unsigned long long)(uintptr_t)gsrc;
  u32x4 g0;
  g0[0] = 1u;                                    // count=1 valid descriptor
  g0[1] = lds_off_bytes;                         // LDS byte address
  g0[2] = (unsigned)(ga & 0xffffffffu);          // global_addr[31:0]
  g0[3] = (unsigned)((ga >> 32) & 0x1ffffffu) | (2u << 30);  // addr[56:32]|type=2
  i32x8 g1;
  g1[0] = (int)((1u << 16) | (1u << 20) |
                (pad_interval_code << 22) | (pad_amount_code << 25));
  g1[1] = (int)(tile_w_elems << 16);             // tensor_dim0[15:0]
  g1[2] = (int)(tile_rows << 16);                // tensor_dim1[15:0]
  g1[3] = (int)(tile_w_elems << 16);             // tile_dim0
  g1[4] = (int)tile_rows;                        // tile_dim1 (tile_dim2=0)
  g1[5] = (int)row_stride_elems;                 // tensor_dim0_stride[31:0]
  g1[6] = 0;
  g1[7] = 0;
  i32x4 z4 = {0, 0, 0, 0};
#if __clang_major__ >= 23
  i32x8 z8 = {0, 0, 0, 0, 0, 0, 0, 0};
  __builtin_amdgcn_tensor_load_to_lds(g0, g1, z4, z4, z8, 0);
#else
  __builtin_amdgcn_tensor_load_to_lds(g0, g1, z4, z4, 0);
#endif
}
#endif

// ---------------------------------------------------------------------------
// LDS fragment loaders (wave32; A/B use contiguous k, not the ISA table order; C/D per ISA)
//   A 16x32 f16: lane m=l&15; elems j: k = (l>>4)*16 + j
//   B 32x16 f16: lane n=l&15; elems j: k = (l>>4)*16 + j
//   C/D 16x16 f32: VGPR r: m = (l>>4)*8 + r, n = l&15
// ---------------------------------------------------------------------------
__device__ __forceinline__ v16h lds_a_frag(const half_t* base, int pitch, int m,
                                           int koff, int lane) {
  const half_t* p = base + m * pitch + koff + ((lane >> 4) << 4);
  v8h lo = *(const v8h*)p;
  v8h hi = *(const v8h*)(p + 8);
  return __builtin_shufflevector(lo, hi, 0, 1, 2, 3, 4, 5, 6, 7,
                                 8, 9, 10, 11, 12, 13, 14, 15);
}

// B staged n-major in LDS: sB[n][k]
__device__ __forceinline__ v16h lds_b_frag_nmajor(const half_t* base, int pitch, int n,
                                                  int koff, int lane) {
  const half_t* p = base + n * pitch + koff + ((lane >> 4) << 4);
  v8h lo = *(const v8h*)p;
  v8h hi = *(const v8h*)(p + 8);
  return __builtin_shufflevector(lo, hi, 0, 1, 2, 3, 4, 5, 6, 7,
                                 8, 9, 10, 11, 12, 13, 14, 15);
}

// B staged k-major in LDS: sB[k][n]
__device__ __forceinline__ v16h lds_b_frag_kmajor(const half_t* base, int pitch, int n,
                                                  int koff, int lane) {
  v16h r;
  int k0 = koff + ((lane >> 4) << 4);
#pragma unroll
  for (int j = 0; j < 16; ++j) r[j] = base[(k0 + j) * pitch + n];
  return r;
}

__device__ __forceinline__ float gelu_exact(float x) {
  return 0.5f * x * (1.0f + erff(x * 0.70710678118654752f));
}

// ---------------------------------------------------------------------------
// Generic WMMA GEMM: C[M,N] = op(A[M,K] @ B + bias)
//   BT=false : B is row-major [K,N];  BT=true : B source is [N,K] (A @ B^T)
//   AHALF    : A is f16, else f32 (converted to f16 while staging)
// Epilogues: 0 store f32 | 1 store f16 | 2 bias+GELU store f16 | 3 bias store f32
// Block tile 128x128, BK=32; 8 waves in 4x2 grid, 32x64/wave (8 WMMA/k-step).
// ---------------------------------------------------------------------------
template <bool BT, bool AHALF, int EPI>
__global__ __launch_bounds__(256) void wmma_gemm_kernel(
    const void* __restrict__ Av, const float* __restrict__ Bsrc,
    const float* __restrict__ bias, void* __restrict__ Cv,
    int M, int N, int K, int lda, int ldb, int ldc) {
  constexpr int BM = 128, BN = 128, BK = 32, PA = 40, PB = 40;
  __shared__ alignas(16) half_t sA[BM * PA];
  __shared__ alignas(16) half_t sB[BN * PB];

  const int t = threadIdx.x;
  const int wave = t >> 5, lane = t & 31;
  const int m0 = blockIdx.y * BM, n0 = blockIdx.x * BN;
  const int wm = (wave >> 1) * 32, wn = (wave & 1) * 64;

  const v8f vzero = {0.f, 0.f, 0.f, 0.f, 0.f, 0.f, 0.f, 0.f};
  v8f acc[2][4];
#pragma unroll
  for (int mi = 0; mi < 2; ++mi)
#pragma unroll
    for (int ni = 0; ni < 4; ++ni) acc[mi][ni] = vzero;

  for (int k0 = 0; k0 < K; k0 += BK) {
    // --- stage A tile (128 x 32) into LDS as f16, row-major [m][k] ---
    if constexpr (AHALF) {
      const half_t* A = (const half_t*)Av;
#pragma unroll
      for (int p = 0; p < 4; ++p) {
        int row = p * 32 + (t >> 3);
        int c4  = (t & 7) * 4;
        v4h v = *(const v4h*)(A + (size_t)(m0 + row) * lda + k0 + c4);
        *(v4h*)&sA[row * PA + c4] = v;
      }
    } else {
      const float* A = (const float*)Av;
#pragma unroll
      for (int p = 0; p < 4; ++p) {
        int row = p * 32 + (t >> 3);
        int c4  = (t & 7) * 4;
        v4f v = *(const v4f*)(A + (size_t)(m0 + row) * lda + k0 + c4);
        v4h h;
        h.x = (half_t)v.x; h.y = (half_t)v.y; h.z = (half_t)v.z; h.w = (half_t)v.w;
        *(v4h*)&sA[row * PA + c4] = h;
      }
    }
    // --- stage B tile (32k x 128n) into LDS as f16, n-major [n][k] ---
    if constexpr (BT) {
#pragma unroll
      for (int p = 0; p < 4; ++p) {
        int idx = p * 256 + t;
        int n   = idx >> 3;
        int c4  = (idx & 7) * 4;
        v4f v = *(const v4f*)(Bsrc + (size_t)(n0 + n) * ldb + k0 + c4);
        v4h h;
        h.x = (half_t)v.x; h.y = (half_t)v.y; h.z = (half_t)v.z; h.w = (half_t)v.w;
        *(v4h*)&sB[n * PB + c4] = h;
      }
    } else {
#pragma unroll
      for (int p = 0; p < 4; ++p) {
        int idx = p * 256 + t;
        int k   = idx >> 5;
        int n4  = (idx & 31) * 4;
        v4f v = *(const v4f*)(Bsrc + (size_t)(k0 + k) * ldb + n0 + n4);
        sB[(n4 + 0) * PB + k] = (half_t)v.x;
        sB[(n4 + 1) * PB + k] = (half_t)v.y;
        sB[(n4 + 2) * PB + k] = (half_t)v.z;
        sB[(n4 + 3) * PB + k] = (half_t)v.w;
      }
    }
    // prefetch next A tile rows into cache (global_prefetch_b8 path)
    if (k0 + BK < K) {
      if constexpr (AHALF)
        __builtin_prefetch((const half_t*)Av + (size_t)(m0 + (t >> 1)) * lda + k0 + BK, 0, 1);
      else
        __builtin_prefetch((const float*)Av + (size_t)(m0 + (t >> 1)) * lda + k0 + BK, 0, 1);
    }
    __syncthreads();

    v16h a0 = lds_a_frag(sA, PA, wm + (lane & 15), 0, lane);
    v16h a1 = lds_a_frag(sA, PA, wm + 16 + (lane & 15), 0, lane);
    v16h b0 = lds_b_frag_nmajor(sB, PB, wn + (lane & 15), 0, lane);
    v16h b1 = lds_b_frag_nmajor(sB, PB, wn + 16 + (lane & 15), 0, lane);
    v16h b2 = lds_b_frag_nmajor(sB, PB, wn + 32 + (lane & 15), 0, lane);
    v16h b3 = lds_b_frag_nmajor(sB, PB, wn + 48 + (lane & 15), 0, lane);

    acc[0][0] = WMMA_F16(a0, b0, acc[0][0]);
    acc[0][1] = WMMA_F16(a0, b1, acc[0][1]);
    acc[0][2] = WMMA_F16(a0, b2, acc[0][2]);
    acc[0][3] = WMMA_F16(a0, b3, acc[0][3]);
    acc[1][0] = WMMA_F16(a1, b0, acc[1][0]);
    acc[1][1] = WMMA_F16(a1, b1, acc[1][1]);
    acc[1][2] = WMMA_F16(a1, b2, acc[1][2]);
    acc[1][3] = WMMA_F16(a1, b3, acc[1][3]);
    __syncthreads();
  }

  // --- epilogue ---
#pragma unroll
  for (int mi = 0; mi < 2; ++mi) {
#pragma unroll
    for (int ni = 0; ni < 4; ++ni) {
      const int col = n0 + wn + ni * 16 + (lane & 15);
      float bv = 0.f;
      if constexpr (EPI == 2 || EPI == 3) bv = bias[col];
#pragma unroll
      for (int r = 0; r < 8; ++r) {
        const int row = m0 + wm + mi * 16 + ((lane >> 4) << 3) + r;
        float v = acc[mi][ni][r];
        if constexpr (EPI == 2) v = gelu_exact(v + bv);
        if constexpr (EPI == 3) v = v + bv;
        if constexpr (EPI == 1 || EPI == 2)
          ((half_t*)Cv)[(size_t)row * ldc + col] = (half_t)v;
        else
          ((float*)Cv)[(size_t)row * ldc + col] = v;
      }
    }
  }
}

// ---------------------------------------------------------------------------
// ProbSparse sparsity metric: M[b,h,l] = max_s(q.k_s) - sum_s(q.k_s)/L
// ---------------------------------------------------------------------------
__global__ __launch_bounds__(256) void probs_m_kernel(
    const half_t* __restrict__ Qh, const half_t* __restrict__ Kh,
    const int* __restrict__ sidx, float* __restrict__ Mbuf) {
  const int tid = blockIdx.x * 256 + threadIdx.x;
  const int bh = tid >> 11, l = tid & (Lc - 1);
  const int b = bh >> 3, h = bh & 7;

  const half_t* q = Qh + ((size_t)(b * Lc + l)) * Dc + h * DKc;
  float qr[64];
#pragma unroll
  for (int i = 0; i < 8; ++i) {
    v8h v = *(const v8h*)(q + i * 8);
#pragma unroll
    for (int j = 0; j < 8; ++j) qr[i * 8 + j] = (float)v[j];
  }
  float mx = -3.4e38f, sm = 0.f;
  for (int s = 0; s < SK; ++s) {
    const int idx = sidx[l * SK + s];
    const half_t* kp = Kh + ((size_t)(b * Lc + idx)) * Dc + h * DKc;
    float d = 0.f;
#pragma unroll
    for (int i = 0; i < 8; ++i) {
      v8h v = *(const v8h*)(kp + i * 8);
#pragma unroll
      for (int j = 0; j < 8; ++j) d += qr[i * 8 + j] * (float)v[j];
    }
    mx = fmaxf(mx, d);
    sm += d;
  }
  Mbuf[tid] = mx - sm * (1.0f / (float)Lc);
}

// ---------------------------------------------------------------------------
// Top-40 selection per (b,h): iterative argmax over L=2048 with LDS reduction.
// ---------------------------------------------------------------------------
__global__ __launch_bounds__(256) void topk_kernel(const float* __restrict__ Mbuf,
                                                   int* __restrict__ top) {
  __shared__ float sv[Lc];
  __shared__ float rv[256];
  __shared__ int   ri[256];
  const int bh = blockIdx.x, t = threadIdx.x;
  for (int i = t; i < Lc; i += 256) sv[i] = Mbuf[bh * Lc + i];
  __syncthreads();
  for (int it = 0; it < SK; ++it) {
    float bv = -3.4e38f;
    int bi = Lc;
    for (int i = t; i < Lc; i += 256) {
      float v = sv[i];
      if (v > bv || (v == bv && i < bi)) { bv = v; bi = i; }
    }
    rv[t] = bv; ri[t] = bi;
    __syncthreads();
    for (int s = 128; s > 0; s >>= 1) {
      if (t < s) {
        if (rv[t + s] > rv[t] || (rv[t + s] == rv[t] && ri[t + s] < ri[t])) {
          rv[t] = rv[t + s]; ri[t] = ri[t + s];
        }
      }
      __syncthreads();
    }
    if (t == 0) { top[bh * SK + it] = ri[0]; sv[ri[0]] = -3.4e38f; }
    __syncthreads();
  }
}

// Gather Q_red (padded to 48 rows, pad rows = 0) as f16 [bh][48][64].
__global__ __launch_bounds__(256) void gather_qred_kernel(
    const half_t* __restrict__ Qh, const int* __restrict__ top,
    half_t* __restrict__ Qred) {
  const int bh = blockIdx.x, b = bh >> 3, h = bh & 7;
  for (int e = threadIdx.x; e < UP * DKc; e += 256) {
    const int u = e >> 6, d = e & 63;
    half_t v = (half_t)0.f;
    if (u < SK) {
      const int l = top[bh * SK + u];
      v = Qh[((size_t)(b * Lc + l)) * Dc + h * DKc + d];
    }
    Qred[(size_t)bh * UP * DKc + e] = v;
  }
}

// ---------------------------------------------------------------------------
// scores[bh, 0..47, l] = (Q_red @ K^T) / 8   (WMMA, M=48, K=64, N-chunk=128)
// TDM stages both the Q_red tile (64x48) and the K tile (64x128, stride 512).
// ---------------------------------------------------------------------------
__global__ __launch_bounds__(256) void scores_kernel(
    const half_t* __restrict__ Qred, const half_t* __restrict__ Kh,
    float* __restrict__ scores) {
  constexpr int P = 72;
  __shared__ alignas(16) half_t sQ[UP * P];
  __shared__ alignas(16) half_t sK[128 * P];
  const int t = threadIdx.x, lane = t & 31, wave = t >> 5;
  const int bh = blockIdx.y, b = bh >> 3, h = bh & 7;
  const int n0 = blockIdx.x * 128;

#if USE_TDM
  if (wave == 0) {
    tdm_load_2d_f16((unsigned)(uintptr_t)&sQ[0],
                    Qred + (size_t)bh * UP * DKc,
                    64, UP, 64, /*32 DW rows*/ 4, /*pad 4 DW*/ 3);
    tdm_load_2d_f16((unsigned)(uintptr_t)&sK[0],
                    Kh + ((size_t)(b * Lc + n0)) * Dc + h * DKc,
                    64, 128, Dc, /*32 DW rows*/ 4, /*pad 4 DW*/ 3);
    __builtin_amdgcn_s_wait_tensorcnt(0);
  }
#else
  for (int e = t; e < UP * DKc; e += 256) {
    const int u = e >> 6, d = e & 63;
    sQ[u * P + d] = Qred[(size_t)bh * UP * DKc + e];
  }
#pragma unroll
  for (int p = 0; p < 4; ++p) {
    const int idx = p * 256 + t;
    const int n = idx >> 3, kk = (idx & 7) * 8;
    v8h v = *(const v8h*)(Kh + ((size_t)(b * Lc + n0 + n)) * Dc + h * DKc + kk);
    *(v8h*)&sK[n * P + kk] = v;
  }
#endif
  __syncthreads();

  const v8f vzero = {0.f, 0.f, 0.f, 0.f, 0.f, 0.f, 0.f, 0.f};
  v8f acc[3];
  acc[0] = vzero; acc[1] = vzero; acc[2] = vzero;
  const int wn = wave * 16;
#pragma unroll
  for (int ks = 0; ks < 2; ++ks) {
    v16h bf = lds_b_frag_nmajor(sK, P, wn + (lane & 15), ks * 32, lane);
#pragma unroll
    for (int mi = 0; mi < 3; ++mi) {
      v16h af = lds_a_frag(sQ, P, mi * 16 + (lane & 15), ks * 32, lane);
      acc[mi] = WMMA_F16(af, bf, acc[mi]);
    }
  }
  const int col = n0 + wn + (lane & 15);
#pragma unroll
  for (int mi = 0; mi < 3; ++mi)
#pragma unroll
    for (int r = 0; r < 8; ++r) {
      const int row = mi * 16 + ((lane >> 4) << 3) + r;
      scores[(size_t)bh * UP * Lc + (size_t)row * Lc + col] = acc[mi][r] * 0.125f;
    }
}

// Row softmax over L=2048; pad rows (u>=40) written as zeros. Output f16.
__global__ __launch_bounds__(256) void softmax_kernel(
    const float* __restrict__ scores, half_t* __restrict__ attn) {
  const int u = blockIdx.x, bh = blockIdx.y, t = threadIdx.x;
  const size_t base = ((size_t)bh * UP + u) * Lc;
  __shared__ float red[256];
  if (u >= SK) {
    for (int i = t; i < Lc; i += 256) attn[base + i] = (half_t)0.f;
    return;
  }
  float v[8];
  float mx = -3.4e38f;
#pragma unroll
  for (int i = 0; i < 8; ++i) { v[i] = scores[base + t + i * 256]; mx = fmaxf(mx, v[i]); }
  red[t] = mx; __syncthreads();
  for (int s = 128; s > 0; s >>= 1) { if (t < s) red[t] = fmaxf(red[t], red[t + s]); __syncthreads(); }
  mx = red[0]; __syncthreads();
  float sm = 0.f;
#pragma unroll
  for (int i = 0; i < 8; ++i) { v[i] = expf(v[i] - mx); sm += v[i]; }
  red[t] = sm; __syncthreads();
  for (int s = 128; s > 0; s >>= 1) { if (t < s) red[t] += red[t + s]; __syncthreads(); }
  const float inv = 1.f / red[0];
#pragma unroll
  for (int i = 0; i < 8; ++i) attn[base + t + i * 256] = (half_t)(v[i] * inv);
}

// V column means per (b,h): vmean[b, h*64+d]
__global__ __launch_bounds__(256) void vmean_kernel(const half_t* __restrict__ Vh,
                                                    float* __restrict__ vmean) {
  const int bh = blockIdx.x, b = bh >> 3, h = bh & 7;
  const int t = threadIdx.x, d = t & 63, part = t >> 6;
  __shared__ float red[256];
  float s = 0.f;
  for (int l = part * 512; l < (part + 1) * 512; ++l)
    s += (float)Vh[((size_t)(b * Lc + l)) * Dc + h * DKc + d];
  red[t] = s; __syncthreads();
  if (t < 64) {
    const float tot = red[t] + red[t + 64] + red[t + 128] + red[t + 192];
    vmean[b * Dc + h * DKc + t] = tot * (1.0f / (float)Lc);
  }
}

// Broadcast-fill ctx rows with the per-(b,h) V mean (f16).
__global__ __launch_bounds__(256) void fill_ctx_kernel(const float* __restrict__ vmean,
                                                       half_t* __restrict__ ctxh) {
  const size_t idx8 = ((size_t)blockIdx.x * 256 + threadIdx.x) * 8;
  const int n = (int)(idx8 & (Dc - 1));
  const int m = (int)(idx8 >> 9);
  const int b = m >> 11;
#pragma unroll
  for (int i = 0; i < 8; ++i) ctxh[idx8 + i] = (half_t)vmean[b * Dc + n + i];
}

// ---------------------------------------------------------------------------
// Attention update: upd = attn[48x2048] @ V[2048x64] (WMMA, waves split K),
// cross-wave LDS ds_add_f32 reduction, scatter real rows into ctx at top-k.
// TDM stages the attn chunk (48x256, stride 2048) and V chunk (256x64, str 512).
// ---------------------------------------------------------------------------
__global__ __launch_bounds__(256) void upd_kernel(
    const half_t* __restrict__ attn, const half_t* __restrict__ Vh,
    const int* __restrict__ top, half_t* __restrict__ ctxh) {
  constexpr int PA2 = 264, PB2 = 72;
  __shared__ alignas(16) half_t sAa[UP * PA2];
  __shared__ alignas(16) half_t sBb[256 * PB2];
  __shared__ float sC[UP * DKc];
  const int t = threadIdx.x, lane = t & 31, wave = t >> 5;
  const int bh = blockIdx.x, b = bh >> 3, h = bh & 7;

  for (int e = t; e < UP * DKc; e += 256) sC[e] = 0.f;

  const v8f vzero = {0.f, 0.f, 0.f, 0.f, 0.f, 0.f, 0.f, 0.f};
  v8f acc[3][4];
#pragma unroll
  for (int mi = 0; mi < 3; ++mi)
#pragma unroll
    for (int ni = 0; ni < 4; ++ni) acc[mi][ni] = vzero;

  for (int kc = 0; kc < Lc; kc += 256) {
    __syncthreads();
#if USE_TDM
    if (wave == 0) {
      // attn rows: 256 halves = 128 DWORDs per row -> pad_interval code 6
      tdm_load_2d_f16((unsigned)(uintptr_t)&sAa[0],
                      attn + (size_t)bh * UP * Lc + kc,
                      256, UP, Lc, /*128 DW rows*/ 6, /*pad 4 DW*/ 3);
      // V rows: 64 halves = 32 DWORDs per row -> pad_interval code 4
      tdm_load_2d_f16((unsigned)(uintptr_t)&sBb[0],
                      Vh + ((size_t)(b * Lc + kc)) * Dc + h * DKc,
                      64, 256, Dc, /*32 DW rows*/ 4, /*pad 4 DW*/ 3);
      __builtin_amdgcn_s_wait_tensorcnt(0);
    }
#else
#pragma unroll
    for (int p = 0; p < 6; ++p) {
      const int idx = p * 256 + t;
      const int u = idx >> 5, kk = (idx & 31) * 8;
      v8h v = *(const v8h*)(attn + ((size_t)bh * UP + u) * Lc + kc + kk);
      *(v8h*)&sAa[u * PA2 + kk] = v;
    }
#pragma unroll
    for (int p = 0; p < 8; ++p) {
      const int idx = p * 256 + t;
      const int kk = idx >> 3, dd = (idx & 7) * 8;
      v8h v = *(const v8h*)(Vh + ((size_t)(b * Lc + kc + kk)) * Dc + h * DKc + dd);
      *(v8h*)&sBb[kk * PB2 + dd] = v;
    }
#endif
    __syncthreads();

    const int kb = wave * 32;
    v16h af[3], bf[4];
#pragma unroll
    for (int mi = 0; mi < 3; ++mi)
      af[mi] = lds_a_frag(sAa, PA2, mi * 16 + (lane & 15), kb, lane);
#pragma unroll
    for (int ni = 0; ni < 4; ++ni)
      bf[ni] = lds_b_frag_kmajor(sBb, PB2, ni * 16 + (lane & 15), kb, lane);
#pragma unroll
    for (int mi = 0; mi < 3; ++mi)
#pragma unroll
      for (int ni = 0; ni < 4; ++ni) acc[mi][ni] = WMMA_F16(af[mi], bf[ni], acc[mi][ni]);
  }
  __syncthreads();

#pragma unroll
  for (int mi = 0; mi < 3; ++mi)
#pragma unroll
    for (int ni = 0; ni < 4; ++ni) {
      const int col = ni * 16 + (lane & 15);
#pragma unroll
      for (int r = 0; r < 8; ++r) {
        const int row = mi * 16 + ((lane >> 4) << 3) + r;
        atomicAdd(&sC[row * DKc + col], acc[mi][ni][r]);
      }
    }
  __syncthreads();

  for (int e = t; e < SK * DKc; e += 256) {
    const int u = e >> 6, d = e & 63;
    const int l = top[bh * SK + u];
    ctxh[((size_t)(b * Lc + l)) * Dc + h * DKc + d] = (half_t)sC[u * DKc + d];
  }
}

// Residual add + LayerNorm over D=512 (one block per row).
__global__ __launch_bounds__(256) void add_ln_kernel(
    const float* __restrict__ a, const float* __restrict__ r,
    const float* __restrict__ g, const float* __restrict__ bb,
    float* __restrict__ out) {
  const int row = blockIdx.x, t = threadIdx.x;
  __shared__ float red[256];
  const size_t base = (size_t)row * Dc;
  const float s0 = a[base + t] + r[base + t];
  const float s1 = a[base + t + 256] + r[base + t + 256];
  red[t] = s0 + s1; __syncthreads();
  for (int s = 128; s > 0; s >>= 1) { if (t < s) red[t] += red[t + s]; __syncthreads(); }
  const float mean = red[0] * (1.0f / (float)Dc); __syncthreads();
  const float d0 = s0 - mean, d1 = s1 - mean;
  red[t] = d0 * d0 + d1 * d1; __syncthreads();
  for (int s = 128; s > 0; s >>= 1) { if (t < s) red[t] += red[t + s]; __syncthreads(); }
  const float inv = rsqrtf(red[0] * (1.0f / (float)Dc) + 1e-5f);
  out[base + t]       = d0 * inv * g[t] + bb[t];
  out[base + t + 256] = d1 * inv * g[t + 256] + bb[t + 256];
}

// ---------------------------------------------------------------------------
// Host-side orchestration (all on `stream`; workspace need ~116 MB)
// ---------------------------------------------------------------------------
extern "C" void kernel_launch(void* const* d_in, const int* in_sizes, int n_in,
                              void* d_out, int out_size, void* d_ws, size_t ws_size,
                              hipStream_t stream) {
  const float* x       = (const float*)d_in[0];
  const float* Wq      = (const float*)d_in[1];
  const float* Wk      = (const float*)d_in[2];
  const float* Wv      = (const float*)d_in[3];
  const float* Wo      = (const float*)d_in[4];
  const float* ln1_g   = (const float*)d_in[5];
  const float* ln1_b   = (const float*)d_in[6];
  const float* conv1_w = (const float*)d_in[7];
  const float* conv1_b = (const float*)d_in[8];
  const float* conv2_w = (const float*)d_in[9];
  const float* conv2_b = (const float*)d_in[10];
  const float* ln2_g   = (const float*)d_in[11];
  const float* ln2_b   = (const float*)d_in[12];
  const int*   sidx    = (const int*)d_in[13];

  char* ws = (char*)d_ws;
  const size_t MB = 1ull << 20;
  half_t* Qh    = (half_t*)(ws + 0 * MB);    // 8 MB  [8192,512] f16
  half_t* Kh    = (half_t*)(ws + 8 * MB);    // 8 MB
  half_t* Vh    = (half_t*)(ws + 16 * MB);   // 8 MB
  half_t* ctxh  = (half_t*)(ws + 24 * MB);   // 8 MB
  float*  x1    = (float*)(ws + 32 * MB);    // 16 MB [8192,512] f32
  float*  tmpf  = (float*)(ws + 48 * MB);    // 16 MB (attn-out, later y2)
  half_t* ff1h  = (half_t*)(ws + 64 * MB);   // 32 MB [8192,2048] f16
  char*   small = ws + 96 * MB;
  float*  Mbuf  = (float*)(small);                        // 256 KB
  int*    top   = (int*)(small + 0x40000);                // 5 KB
  half_t* Qred  = (half_t*)(small + 0x42000);             // 192 KB
  float*  sc    = (float*)(small + 0x80000);              // 12.6 MB
  half_t* attnh = (half_t*)(small + 0x80000 + (size_t)32 * UP * Lc * 4); // 6.3 MB
  float*  vmean = (float*)(small + 0x80000 + (size_t)32 * UP * Lc * 4
                                          + (size_t)32 * UP * Lc * 2);   // 8 KB

  const dim3 blk(256);

  // 1) Q/K/V projections: [8192,512] = x @ W (f16 outputs for WMMA reuse)
  wmma_gemm_kernel<false, false, 1><<<dim3(4, 64), blk, 0, stream>>>(
      x, Wq, nullptr, Qh, Mrows, Dc, Dc, Dc, Dc, Dc);
  wmma_gemm_kernel<false, false, 1><<<dim3(4, 64), blk, 0, stream>>>(
      x, Wk, nullptr, Kh, Mrows, Dc, Dc, Dc, Dc, Dc);
  wmma_gemm_kernel<false, false, 1><<<dim3(4, 64), blk, 0, stream>>>(
      x, Wv, nullptr, Vh, Mrows, Dc, Dc, Dc, Dc, Dc);

  // 2) ProbSparse metric, top-k, Q_red gather
  probs_m_kernel<<<(Bc * Hc * Lc) / 256, blk, 0, stream>>>(Qh, Kh, sidx, Mbuf);
  topk_kernel<<<Bc * Hc, blk, 0, stream>>>(Mbuf, top);
  gather_qred_kernel<<<Bc * Hc, blk, 0, stream>>>(Qh, top, Qred);

  // 3) Reduced attention: scores (WMMA+TDM) -> softmax -> update (WMMA+TDM)
  scores_kernel<<<dim3(16, Bc * Hc), blk, 0, stream>>>(Qred, Kh, sc);
  softmax_kernel<<<dim3(UP, Bc * Hc), blk, 0, stream>>>(sc, attnh);
  vmean_kernel<<<Bc * Hc, blk, 0, stream>>>(Vh, vmean);
  fill_ctx_kernel<<<(Mrows * Dc) / (256 * 8), blk, 0, stream>>>(vmean, ctxh);
  upd_kernel<<<Bc * Hc, blk, 0, stream>>>(attnh, Vh, top, ctxh);

  // 4) Output projection + residual LN1
  wmma_gemm_kernel<false, true, 0><<<dim3(4, 64), blk, 0, stream>>>(
      ctxh, Wo, nullptr, tmpf, Mrows, Dc, Dc, Dc, Dc, Dc);
  add_ln_kernel<<<Mrows, blk, 0, stream>>>(x, tmpf, ln1_g, ln1_b, x1);

  // 5) FFN: GELU(x1 @ W1^T + b1) @ W2^T + b2, residual LN2 -> d_out
  wmma_gemm_kernel<true, false, 2><<<dim3(16, 64), blk, 0, stream>>>(
      x1, conv1_w, conv1_b, ff1h, Mrows, DFF, Dc, Dc, Dc, DFF);
  wmma_gemm_kernel<true, true, 3><<<dim3(4, 64), blk, 0, stream>>>(
      ff1h, conv2_w, conv2_b, tmpf, Mrows, Dc, DFF, DFF, DFF, Dc);
  add_ln_kernel<<<Mrows, blk, 0, stream>>>(x1, tmpf, ln2_g, ln2_b, (float*)d_out);
}
